// GAB_21285857919709
// MI455X (gfx1250) — matrix operand order repaired and checked by reading, not compiled, not run on hardware
//
#include <hip/hip_runtime.h>
#include <math.h>

#define HEADS 6
#define HD 16
#define POSD 6
#define NB 48
#define NN 512
#define SMAXCAP 2048
#define HSTR 8
#define POSCAP 256   // LDS copy of the collapsed 1-D bias table (43 rows used for 8x8x8)
#define CHUNK 64     // keys staged per async fill

typedef __attribute__((ext_vector_type(16))) _Float16 v16h;
typedef __attribute__((ext_vector_type(8)))  float    v8f;

// d_ws float-offset layout:
//  [0, 2*SMAXCAP*HSTR)              : pos tables (2 tables, HSTR-strided heads)
//  [CSUM_OFF, CSUM_OFF+512)         : csum[] as int
//  [O1_OFF, ...)                    : pass-1 output (b,H,n,hd) f32
#define POS_OFF  0
#define CSUM_OFF (2 * SMAXCAP * HSTR)
#define O1_OFF   (CSUM_OFF + 1024)

__device__ __forceinline__ void ln6(float* x, const float* g, const float* b) {
    float mu = 0.f;
    #pragma unroll
    for (int i = 0; i < POSD; ++i) mu += x[i];
    mu *= (1.0f / POSD);
    float var = 0.f;
    #pragma unroll
    for (int i = 0; i < POSD; ++i) { float t = x[i] - mu; var += t * t; }
    var *= (1.0f / POSD);
    float inv = rsqrtf(var + 1e-5f);
    #pragma unroll
    for (int i = 0; i < POSD; ++i) x[i] = (x[i] - mu) * inv * g[i] + b[i];
}

// One block. Computes csum[] and the two tiny pos-MLP tables.
__global__ __launch_bounds__(512)
void GAB_prep_kernel(const float* pw, const float* pb,
                     const float* g1, const float* be1, const float* w1, const float* bb1,
                     const float* g2, const float* be2, const float* w2, const float* bb2,
                     const float* g3, const float* be3, const float* w3, const float* bb3,
                     const int* hp, const int* wp, const int* dp, float* ws)
{
    const int h = *hp, w = *wp, d = *dp;
    const int tid = threadIdx.x;
    const int n = h * w * d;

    int* csum = (int*)(ws + CSUM_OFF);
    for (int i = tid; i < n; i += blockDim.x) {
        int ch = i / (w * d);
        int cw = (i / d) % w;
        int cd = i % d;
        csum[i] = ch + cw + cd;
    }

    int smax = 2 * (h + w + d) - 5;
    if (smax > SMAXCAP) smax = SMAXCAP;
    const int W2 = 2 * w - 1, D2 = 2 * d - 1;

    for (int task = tid; task < 2 * smax; task += blockDim.x) {
        const int t = task / smax;   // which table (0/1)
        const int s = task % smax;   // flat bias-row index (== possible idx value)

        float bi[3];
        bi[0] = (float)(s / (W2 * D2) - (h - 1));
        bi[1] = (float)((s / D2) % W2 - (w - 1));
        bi[2] = (float)(s % D2 - (d - 1));

        float x[POSD], y[POSD];
        #pragma unroll
        for (int o = 0; o < POSD; ++o) {
            const float* r = pw + (size_t)(t * POSD + o) * 3;
            x[o] = r[0] * bi[0] + r[1] * bi[1] + r[2] * bi[2] + pb[t * POSD + o];
        }
        ln6(x, g1 + t * POSD, be1 + t * POSD);
        #pragma unroll
        for (int o = 0; o < POSD; ++o) {
            const float* r = w1 + (size_t)(t * POSD + o) * POSD;
            float acc = bb1[t * POSD + o];
            #pragma unroll
            for (int j = 0; j < POSD; ++j) acc += fmaxf(x[j], 0.f) * r[j];
            y[o] = acc;
        }
        ln6(y, g2 + t * POSD, be2 + t * POSD);
        #pragma unroll
        for (int o = 0; o < POSD; ++o) {
            const float* r = w2 + (size_t)(t * POSD + o) * POSD;
            float acc = bb2[t * POSD + o];
            #pragma unroll
            for (int j = 0; j < POSD; ++j) acc += fmaxf(y[j], 0.f) * r[j];
            x[o] = acc;
        }
        ln6(x, g3 + t * POSD, be3 + t * POSD);
        #pragma unroll
        for (int hh = 0; hh < HEADS; ++hh) {
            const float* r = w3 + (size_t)(t * HEADS + hh) * POSD;
            float acc = bb3[t * HEADS + hh];
            #pragma unroll
            for (int j = 0; j < POSD; ++j) acc += fmaxf(x[j], 0.f) * r[j];
            ws[POS_OFF + (size_t)(t * SMAXCAP + s) * HSTR + hh] = acc;
        }
    }
}

// Flash-attention pass. One wave handles 16 query rows of one (b, head).
// 8 waves/block; since 8 | 32 query tiles, a block always maps to ONE (b, head),
// so K/V are staged in LDS once per block via async copies (double-buffered).
__global__ __launch_bounds__(256)
void GAB_flash_kernel(const float* __restrict__ qkv, const float* __restrict__ grid,
                      float* __restrict__ out, float* __restrict__ ws,
                      const int* hp, const int* wp, const int* dp, int pass)
{
    const int tid  = threadIdx.x;
    const int lane = tid & 31;
    const int half = lane >> 4;
    const int lm   = lane & 15;

    const int wpb = blockDim.x >> 5;                 // 8 waves per block
    const int wv  = blockIdx.x * wpb + (tid >> 5);
    const int qt  = wv & 31;                         // NN/16 = 32 query tiles
    const int bh  = (blockIdx.x * wpb) >> 5;         // constant per block
    const int head = bh % HEADS;
    const int b    = bh / HEADS;

    const int off3 = *hp + *wp + *dp - 3;
    const int* csum_g = (const int*)(ws + CSUM_OFF);
    float* o1 = ws + O1_OFF;

    // ---- block-shared staging ----
    __shared__ int      s_csum[NN];
    __shared__ float    s_pos[POSCAP];
    __shared__ float    kbuf[2][CHUNK * HD];
    __shared__ float    vbuf[2][CHUNK * HD];
    __shared__ _Float16 plds[8][16 * 32];

    for (int i = tid; i < NN; i += blockDim.x) s_csum[i] = csum_g[i];
    {
        const float* pv = ws + POS_OFF + (size_t)pass * SMAXCAP * HSTR + head;
        for (int i = tid; i < POSCAP; i += blockDim.x)
            s_pos[i] = pv[(size_t)i * HSTR];
    }

    const float* Qp; int qld;
    const float* Kp; int kld;
    const float* Vp; int vld;
    float* Op; int old_; int ocol;
    if (pass == 0) {   // attn(grid, k, v) + rpb0  -> ws
        Qp = grid + (size_t)b * NN * 96 + head * HD;        qld = 96;
        Kp = qkv  + (size_t)b * NN * 288 + 96 + head * HD;  kld = 288;
        Vp = qkv  + (size_t)b * NN * 288 + 192 + head * HD; vld = 288;
        Op = o1 + (size_t)(b * HEADS + head) * NN * HD;     old_ = HD;  ocol = 0;
    } else {           // attn(q, grid, o1) + rpb1 -> out
        Qp = qkv  + (size_t)b * NN * 288 + head * HD;       qld = 288;
        Kp = grid + (size_t)b * NN * 96 + head * HD;        kld = 96;
        Vp = o1 + (size_t)(b * HEADS + head) * NN * HD;     vld = HD;
        Op = out + (size_t)b * NN * 96;                     old_ = 96; ocol = head * HD;
    }
    const float scale = 0.25f;  // HD^-0.5

    const int qrow0 = qt * 16;

    // ---- Q tile -> 16-bit A-operand layout (K padded 16 -> 32) ----
    v16h aQ;
    {
        const float* qr = Qp + (size_t)(qrow0 + lm) * qld;
        if (!half) {
        const float4 q0 = *(const float4*)(qr);
        const float4 q1 = *(const float4*)(qr + 4);
        const float4 q2 = *(const float4*)(qr + 8);
        const float4 q3 = *(const float4*)(qr + 12);
        aQ[0] = (_Float16)(q0.x * scale); aQ[1] = (_Float16)(q0.y * scale);
        aQ[2] = (_Float16)(q0.z * scale); aQ[3] = (_Float16)(q0.w * scale);
        aQ[4] = (_Float16)(q1.x * scale); aQ[5] = (_Float16)(q1.y * scale);
        aQ[6] = (_Float16)(q1.z * scale); aQ[7] = (_Float16)(q1.w * scale);
        aQ[8] = (_Float16)(q2.x * scale); aQ[9] = (_Float16)(q2.y * scale);
        aQ[10] = (_Float16)(q2.z * scale); aQ[11] = (_Float16)(q2.w * scale);
        aQ[12] = (_Float16)(q3.x * scale); aQ[13] = (_Float16)(q3.y * scale);
        aQ[14] = (_Float16)(q3.z * scale); aQ[15] = (_Float16)(q3.w * scale);
        } else {
        #pragma unroll
        for (int j = 0; j < 16; ++j) aQ[j] = (_Float16)0.f;
        }
    }

    v16h bOnes;
    #pragma unroll
    for (int j = 0; j < 16; ++j) bOnes[j] = (_Float16)1.0f;

    // ---- async stage of first K/V chunk (1x b128 per thread per matrix) ----
    const int srow  = tid >> 2;         // 0..63 : key row within chunk
    const int spart = (tid & 3) * 4;    // float offset within row: 0,4,8,12
    {
        unsigned kdst = (unsigned)(size_t)(&kbuf[0][srow * HD + spart]);
        unsigned long long kga = (unsigned long long)(size_t)(Kp + (size_t)srow * kld + spart);
        asm volatile("global_load_async_to_lds_b128 %0, %1, off" :: "v"(kdst), "v"(kga) : "memory");
        unsigned vdst = (unsigned)(size_t)(&vbuf[0][srow * HD + spart]);
        unsigned long long vga = (unsigned long long)(size_t)(Vp + (size_t)srow * vld + spart);
        asm volatile("global_load_async_to_lds_b128 %0, %1, off" :: "v"(vdst), "v"(vga) : "memory");
    }

    int csm[8];
    float mrow[8], lrow[8];
    v8f acc = {};
    _Float16* pl = &plds[tid >> 5][0];

    __syncthreads();   // s_csum / s_pos visible (async chunk waited below)

    #pragma unroll
    for (int r = 0; r < 8; ++r) {
        csm[r] = s_csum[qrow0 + r + 8 * half] + off3;
        mrow[r] = -3.0e38f;
        lrow[r] = 0.f;
    }

    for (int c = 0; c < NN / CHUNK; ++c) {
        // prefetch next chunk into the other buffer, then wait for our chunk
        if (c + 1 < NN / CHUNK) {
            const int nb = (c + 1) & 1;
            const int j0n = (c + 1) * CHUNK;
            unsigned kdst = (unsigned)(size_t)(&kbuf[nb][srow * HD + spart]);
            unsigned long long kga = (unsigned long long)(size_t)(Kp + (size_t)(j0n + srow) * kld + spart);
            asm volatile("global_load_async_to_lds_b128 %0, %1, off" :: "v"(kdst), "v"(kga) : "memory");
            unsigned vdst = (unsigned)(size_t)(&vbuf[nb][srow * HD + spart]);
            unsigned long long vga = (unsigned long long)(size_t)(Vp + (size_t)(j0n + srow) * vld + spart);
            asm volatile("global_load_async_to_lds_b128 %0, %1, off" :: "v"(vdst), "v"(vga) : "memory");
            asm volatile("s_wait_asynccnt 0x2" ::: "memory");   // our chunk done, prefetch in flight
        } else {
            asm volatile("s_wait_asynccnt 0x0" ::: "memory");
        }
        __syncthreads();   // whole block sees chunk c

        const float* kt = kbuf[c & 1];
        const float* vt = vbuf[c & 1];
        const int j0g = c * CHUNK;

        for (int jj = 0; jj < CHUNK; jj += 32) {
            // ---- K from LDS -> two B-operands (K-dim = hd padded to 32) ----
            v16h bK0, bK1;
            if (!half) {
                const float* kr0 = kt + (jj + lm) * HD;
                const float* kr1 = kt + (jj + 16 + lm) * HD;
                const float4 a0 = *(const float4*)(kr0);
                const float4 a1 = *(const float4*)(kr0 + 4);
                const float4 a2 = *(const float4*)(kr0 + 8);
                const float4 a3 = *(const float4*)(kr0 + 12);
                const float4 c0 = *(const float4*)(kr1);
                const float4 c1 = *(const float4*)(kr1 + 4);
                const float4 c2 = *(const float4*)(kr1 + 8);
                const float4 c3 = *(const float4*)(kr1 + 12);
                bK0[0]=(_Float16)a0.x; bK0[1]=(_Float16)a0.y; bK0[2]=(_Float16)a0.z; bK0[3]=(_Float16)a0.w;
                bK0[4]=(_Float16)a1.x; bK0[5]=(_Float16)a1.y; bK0[6]=(_Float16)a1.z; bK0[7]=(_Float16)a1.w;
                bK0[8]=(_Float16)a2.x; bK0[9]=(_Float16)a2.y; bK0[10]=(_Float16)a2.z; bK0[11]=(_Float16)a2.w;
                bK0[12]=(_Float16)a3.x; bK0[13]=(_Float16)a3.y; bK0[14]=(_Float16)a3.z; bK0[15]=(_Float16)a3.w;
                bK1[0]=(_Float16)c0.x; bK1[1]=(_Float16)c0.y; bK1[2]=(_Float16)c0.z; bK1[3]=(_Float16)c0.w;
                bK1[4]=(_Float16)c1.x; bK1[5]=(_Float16)c1.y; bK1[6]=(_Float16)c1.z; bK1[7]=(_Float16)c1.w;
                bK1[8]=(_Float16)c2.x; bK1[9]=(_Float16)c2.y; bK1[10]=(_Float16)c2.z; bK1[11]=(_Float16)c2.w;
                bK1[12]=(_Float16)c3.x; bK1[13]=(_Float16)c3.y; bK1[14]=(_Float16)c3.z; bK1[15]=(_Float16)c3.w;
            } else {
                #pragma unroll
                for (int j = 0; j < 16; ++j) { bK0[j] = (_Float16)0.f; bK1[j] = (_Float16)0.f; }
            }

            v8f z = {};
            v8f s0 = __builtin_amdgcn_wmma_f32_16x16x32_f16(false, aQ, false, bK0, (short)0, z, false, false);
            v8f s1 = __builtin_amdgcn_wmma_f32_16x16x32_f16(false, aQ, false, bK1, (short)0, z, false, false);

            // ---- relative position bias: 1-D table gather from LDS ----
            const int cs0 = s_csum[j0g + jj + lm];
            const int cs1 = s_csum[j0g + jj + 16 + lm];
            #pragma unroll
            for (int r = 0; r < 8; ++r) {
                int i0 = csm[r] - cs0; i0 = (i0 < 0) ? 0 : ((i0 > POSCAP - 1) ? POSCAP - 1 : i0);
                int i1 = csm[r] - cs1; i1 = (i1 < 0) ? 0 : ((i1 > POSCAP - 1) ? POSCAP - 1 : i1);
                s0[r] += s_pos[i0];
                s1[r] += s_pos[i1];
            }

            // ---- online softmax: max via half-wave shuffles, sum via WMMA ----
            float p0[8], p1[8], alpha[8];
            #pragma unroll
            for (int r = 0; r < 8; ++r) {
                float tv = fmaxf(s0[r], s1[r]);
                tv = fmaxf(tv, __shfl_xor(tv, 1, 32));
                tv = fmaxf(tv, __shfl_xor(tv, 2, 32));
                tv = fmaxf(tv, __shfl_xor(tv, 4, 32));
                tv = fmaxf(tv, __shfl_xor(tv, 8, 32));
                const float mnew = fmaxf(mrow[r], tv);
                alpha[r] = __expf(mrow[r] - mnew);
                p0[r] = __expf(s0[r] - mnew);
                p1[r] = __expf(s1[r] - mnew);
                mrow[r] = mnew;
            }

            // ---- P: D-layout -> LDS -> A-layout (same-wave LDS is in-order) ----
            #pragma unroll
            for (int r = 0; r < 8; ++r) {
                const int row = r + 8 * half;
                pl[row * 32 + lm]      = (_Float16)p0[r];
                pl[row * 32 + 16 + lm] = (_Float16)p1[r];
            }
            v16h aP;
            #pragma unroll
            for (int j = 0; j < 8; ++j) {
                const int c0 = (half ? 8 : 0) + ((j < 4) ? 2 * j : 16 + 2 * (j - 4));
                aP[2 * j]     = pl[lm * 32 + c0];
                aP[2 * j + 1] = pl[lm * 32 + c0 + 1];
            }

            // Row sums of P via matrix op (lands exactly where lrow lives).
            v8f sums = __builtin_amdgcn_wmma_f32_16x16x32_f16(false, aP, false, bOnes, (short)0, z, false, false);
            #pragma unroll
            for (int r = 0; r < 8; ++r) {
                lrow[r] = lrow[r] * alpha[r] + sums[r];
                acc[r] *= alpha[r];
            }

            // ---- V from LDS -> B-operand (K-dim = 32 keys, all real) ----
            v16h bV;
            #pragma unroll
            for (int j = 0; j < 8; ++j) {
                const int key = jj + half * 8 + ((j < 4) ? 2 * j : 16 + 2 * (j - 4));
                bV[2 * j]     = (_Float16)vt[key * HD + lm];
                bV[2 * j + 1] = (_Float16)vt[(key + 1) * HD + lm];
            }
            acc = __builtin_amdgcn_wmma_f32_16x16x32_f16(false, aP, false, bV, (short)0, acc, false, false);
        }
        __syncthreads();   // block done reading chunk c; its buffer may be refilled
    }

    // ---- normalize + store ----
    #pragma unroll
    for (int r = 0; r < 8; ++r) {
        const int row = qrow0 + r + 8 * half;
        Op[(size_t)row * old_ + ocol + lm] = acc[r] / lrow[r];
    }
}

extern "C" void kernel_launch(void* const* d_in, const int* in_sizes, int n_in,
                              void* d_out, int out_size, void* d_ws, size_t ws_size,
                              hipStream_t stream) {
    const float* qkv  = (const float*)d_in[0];
    const float* grid = (const float*)d_in[1];
    const float* pw   = (const float*)d_in[2];
    const float* pb   = (const float*)d_in[3];
    const float* g1   = (const float*)d_in[4];
    const float* be1  = (const float*)d_in[5];
    const float* w1   = (const float*)d_in[6];
    const float* bb1  = (const float*)d_in[7];
    const float* g2   = (const float*)d_in[8];
    const float* be2  = (const float*)d_in[9];
    const float* w2   = (const float*)d_in[10];
    const float* bb2  = (const float*)d_in[11];
    const float* g3   = (const float*)d_in[12];
    const float* be3  = (const float*)d_in[13];
    const float* w3   = (const float*)d_in[14];
    const float* bb3  = (const float*)d_in[15];
    const int*   hp   = (const int*)d_in[16];
    const int*   wp   = (const int*)d_in[17];
    const int*   dp   = (const int*)d_in[18];
    float* ws  = (float*)d_ws;
    float* out = (float*)d_out;

    GAB_prep_kernel<<<1, 512, 0, stream>>>(pw, pb, g1, be1, w1, bb1, g2, be2, w2, bb2,
                                           g3, be3, w3, bb3, hp, wp, dp, ws);

    const int waves  = NB * HEADS * (NN / 16);   // 9216
    const int blocks = (waves + 7) / 8;          // 8 waves (256 threads) per block
    GAB_flash_kernel<<<blocks, 256, 0, stream>>>(qkv, grid, out, ws, hp, wp, dp, 0);
    GAB_flash_kernel<<<blocks, 256, 0, stream>>>(qkv, grid, out, ws, hp, wp, dp, 1);
}
